// PianoRollFractalGen_6141803233944
// MI455X (gfx1250) — matrix operand order repaired and checked by reading, not compiled, not run on hardware
//
#include <hip/hip_runtime.h>
#include <hip/hip_bf16.h>

// ---------------------------------------------------------------------------
// ViT encoder (8 blocks) for MI455X / gfx1250.
// GEMMs + attention use v_wmma_f32_16x16x32_f16 (f16 in, fp32 accumulate).
// GEMM tiles staged with async global->LDS (ASYNCcnt) when available.
// ---------------------------------------------------------------------------

typedef __attribute__((ext_vector_type(16))) _Float16 v16h;
typedef __attribute__((ext_vector_type(8)))  _Float16 h8;
typedef __attribute__((ext_vector_type(4)))  _Float16 h4;
typedef __attribute__((ext_vector_type(8)))  float    v8f;
typedef __attribute__((ext_vector_type(4)))  int      i4;

#define TB   4
#define TN   1029
#define TC   1024
#define TH   16
#define TDH  64
#define THID 4096
#define TL   8
#define TM   (TB * TN)          // 4116 rows
#define T3C  (3 * TC)           // 3072
#define ATT_SCALE 0.125f        // 64^-0.5
#define LN_EPS 1e-6f

#if __has_builtin(__builtin_amdgcn_global_load_async_to_lds_b128)
#define USE_ASYNC_LDS 1
#else
#define USE_ASYNC_LDS 0
#endif

#if USE_ASYNC_LDS
#define AS1 __attribute__((address_space(1)))
#define AS3 __attribute__((address_space(3)))
static __device__ __forceinline__ void async_cp16(const _Float16* g,
                                                  _Float16* l) {
  // builtin is typed per payload: b128 -> int4 pointers in AS1 / AS3
  __builtin_amdgcn_global_load_async_to_lds_b128((AS1 i4*)g, (AS3 i4*)l,
                                                 0, 0);
}
static __device__ __forceinline__ void wait_async4() {
#if __has_builtin(__builtin_amdgcn_s_wait_asynccnt)
  __builtin_amdgcn_s_wait_asynccnt(4);
#else
  asm volatile("s_wait_asynccnt 0x4" ::: "memory");
#endif
}
static __device__ __forceinline__ void wait_async0() {
#if __has_builtin(__builtin_amdgcn_s_wait_asynccnt)
  __builtin_amdgcn_s_wait_asynccnt(0);
#else
  asm volatile("s_wait_asynccnt 0x0" ::: "memory");
#endif
}
#endif

static __device__ __forceinline__ v8f wmma_f16(v16h a, v16h b, v8f c) {
  return __builtin_amdgcn_wmma_f32_16x16x32_f16(false, a, false, b,
                                                (short)0, c, false, false);
}

// ---------------------------------------------------------------------------
// fp32 -> f16 conversion (weights / activations)
// ---------------------------------------------------------------------------
__global__ void cvt_f32_f16_kernel(const float* __restrict__ src,
                                   _Float16* __restrict__ dst, int n4) {
  int i = blockIdx.x * blockDim.x + threadIdx.x;
  if (i < n4) {
    float4 f = ((const float4*)src)[i];
    h4 h;
    h.x = (_Float16)f.x; h.y = (_Float16)f.y;
    h.z = (_Float16)f.z; h.w = (_Float16)f.w;
    ((h4*)dst)[i] = h;
  }
}

// ---------------------------------------------------------------------------
// LayerNorm over C=1024; one block (256 thr = 8 waves) per row.
// ---------------------------------------------------------------------------
template <typename OUT_T>
__global__ void ln_kernel(const float* __restrict__ x,
                          const float* __restrict__ w,
                          const float* __restrict__ b,
                          OUT_T* __restrict__ out, int M) {
  int row = blockIdx.x;
  if (row >= M) return;
  int tid = threadIdx.x;
  const float* xr = x + (size_t)row * TC;
  float4 v = ((const float4*)xr)[tid];
  float s  = v.x + v.y + v.z + v.w;
  float ss = v.x * v.x + v.y * v.y + v.z * v.z + v.w * v.w;
#pragma unroll
  for (int o = 16; o > 0; o >>= 1) {
    s  += __shfl_xor(s, o);
    ss += __shfl_xor(ss, o);
  }
  __shared__ float red[2][8];
  int wid = tid >> 5, lane = tid & 31;
  if (lane == 0) { red[0][wid] = s; red[1][wid] = ss; }
  __syncthreads();
  float ts = 0.f, tss = 0.f;
#pragma unroll
  for (int i = 0; i < 8; ++i) { ts += red[0][i]; tss += red[1][i]; }
  float mu = ts * (1.0f / TC);
  float var = tss * (1.0f / TC) - mu * mu;
  float rstd = rsqrtf(var + LN_EPS);
  float4 wv = ((const float4*)w)[tid];
  float4 bv = ((const float4*)b)[tid];
  float o0 = (v.x - mu) * rstd * wv.x + bv.x;
  float o1 = (v.y - mu) * rstd * wv.y + bv.y;
  float o2 = (v.z - mu) * rstd * wv.z + bv.z;
  float o3 = (v.w - mu) * rstd * wv.w + bv.w;
  OUT_T* orow = out + (size_t)row * TC + tid * 4;
  orow[0] = (OUT_T)o0; orow[1] = (OUT_T)o1;
  orow[2] = (OUT_T)o2; orow[3] = (OUT_T)o3;
}

// ---------------------------------------------------------------------------
// WMMA GEMM: out[M,Nout] = A[M,K] * W[Nout,K]^T + bias, fused epilogues.
// Block = 256 thr (8 waves), tile 128x128, K-step 32.  Waves: 4(M) x 2(N),
// each wave owns a 32x64 register tile = 2x4 WMMA accumulators (8 WMMA/step).
// Staging: double-buffered LDS filled by async global->LDS (ASYNCcnt) when
// the toolchain exposes it; else register-staged b128 loads + ds stores.
// ---------------------------------------------------------------------------
template <bool GELU, bool RES, bool OUT16, bool OUT32>
__global__ void gemm_kernel(const _Float16* __restrict__ A,
                            const _Float16* __restrict__ W,
                            const float* __restrict__ bias,
                            const float* __restrict__ res,
                            float* __restrict__ out32,
                            _Float16* __restrict__ out16,
                            int M, int Nout, int K) {
  constexpr int BM = 128, BN = 128, BK = 32, PAD = 8;
#if USE_ASYNC_LDS
  __shared__ _Float16 As[2][BM][BK + PAD];
  __shared__ _Float16 Bs[2][BM][BK + PAD];
#else
  __shared__ _Float16 As[1][BM][BK + PAD];
  __shared__ _Float16 Bs[1][BM][BK + PAD];
#endif

  int tid  = threadIdx.x;
  int wave = tid >> 5;
  int lane = tid & 31;
  int hw   = lane >> 4;          // half-wave 0/1
  int l16  = lane & 15;

  int m_blk = blockIdx.y * BM;
  int n_blk = blockIdx.x * BN;
  int wm = wave & 3;             // 0..3 -> 32-row slab
  int wn = wave >> 2;            // 0..1 -> 64-col slab
  int m_w = m_blk + wm * 32;
  int n_w = n_blk + wn * 64;

  v8f acc[2][4] = {};

  // cooperative-load coordinates: each thread owns 16 halves of A and of B
  int ld_row = tid >> 1, ld_col = (tid & 1) * 16;   // 128 rows x 2 chunks
  int gm = m_blk + ld_row;
  int gn = n_blk + ld_row;
  const _Float16* aptr = A + (size_t)(gm < M ? gm : 0) * K + ld_col;
  const _Float16* wptr = W + (size_t)(gn < Nout ? gn : 0) * K + ld_col;
  bool a_ok = gm < M, b_ok = gn < Nout;

#if USE_ASYNC_LDS
  // -------- async double-buffered pipeline --------
  auto issue_tile = [&](int kb, int buf) {
    if (a_ok) {
      async_cp16(aptr + kb,     &As[buf][ld_row][ld_col]);
      async_cp16(aptr + kb + 8, &As[buf][ld_row][ld_col + 8]);
    }
    if (b_ok) {
      async_cp16(wptr + kb,     &Bs[buf][ld_row][ld_col]);
      async_cp16(wptr + kb + 8, &Bs[buf][ld_row][ld_col + 8]);
    }
  };
  issue_tile(0, 0);
  int buf = 0;
#else
  h8 ar0{}, ar1{}, br0{}, br1{};
  if (a_ok) { ar0 = *(const h8*)(aptr); ar1 = *(const h8*)(aptr + 8); }
  if (b_ok) { br0 = *(const h8*)(wptr); br1 = *(const h8*)(wptr + 8); }
  constexpr int buf = 0;
#endif

  for (int kb = 0; kb < K; kb += BK) {
#if USE_ASYNC_LDS
    if (kb + BK < K) {
      issue_tile(kb + BK, buf ^ 1);       // copy next tile while computing
      wait_async4();                      // oldest 4 (= current tile) done
    } else {
      wait_async0();
    }
    __syncthreads();                      // all waves' tile data visible
#else
    __syncthreads();                       // previous tile's WMMAs done
    *(h8*)&As[0][ld_row][ld_col]     = ar0;
    *(h8*)&As[0][ld_row][ld_col + 8] = ar1;
    *(h8*)&Bs[0][ld_row][ld_col]     = br0;
    *(h8*)&Bs[0][ld_row][ld_col + 8] = br1;
    __syncthreads();
    if (kb + BK < K) {                    // stage next K-tile into registers
      h8 z = {};
      ar0 = z; ar1 = z; br0 = z; br1 = z;
      if (a_ok) {
        ar0 = *(const h8*)(aptr + kb + BK);
        ar1 = *(const h8*)(aptr + kb + BK + 8);
      }
      if (b_ok) {
        br0 = *(const h8*)(wptr + kb + BK);
        br1 = *(const h8*)(wptr + kb + BK + 8);
      }
    }
#endif

    // ---- fragments (16-bit A 16x32 / B 32x16 lane layouts) ----
    v16h af[2];
#pragma unroll
    for (int ai = 0; ai < 2; ++ai) {
      int r = wm * 32 + ai * 16 + l16;
      int k2 = hw * 16;
      h8 lo = *(const h8*)&As[buf][r][k2];
      h8 hi = *(const h8*)&As[buf][r][k2 + 8];
#pragma unroll
      for (int i = 0; i < 8; ++i) { af[ai][i] = lo[i]; af[ai][8 + i] = hi[i]; }
    }
#pragma unroll
    for (int bi = 0; bi < 4; ++bi) {
      int c = wn * 64 + bi * 16 + l16;
      int kk = hw * 16;
      h8 lo = *(const h8*)&Bs[buf][c][kk];
      h8 hi = *(const h8*)&Bs[buf][c][kk + 8];
      v16h bf;
#pragma unroll
      for (int i = 0; i < 8; ++i) { bf[i] = lo[i]; bf[8 + i] = hi[i]; }
      acc[0][bi] = wmma_f16(af[0], bf, acc[0][bi]);
      acc[1][bi] = wmma_f16(af[1], bf, acc[1][bi]);
    }
    __syncthreads();                      // tile consumed; safe to overwrite
#if USE_ASYNC_LDS
    buf ^= 1;
#endif
  }

  // ---- epilogue: C/D layout  m = j + 8*hw,  n = lane&15 ----
#pragma unroll
  for (int ai = 0; ai < 2; ++ai) {
#pragma unroll
    for (int bi = 0; bi < 4; ++bi) {
#pragma unroll
      for (int j = 0; j < 8; ++j) {
        int m = m_w + ai * 16 + j + hw * 8;
        int n = n_w + bi * 16 + l16;
        if (m < M && n < Nout) {
          float v = acc[ai][bi][j] + bias[n];
          if (GELU) v = 0.5f * v * (1.0f + erff(v * 0.70710678118f));
          if (RES)  v += res[(size_t)m * Nout + n];
          if (OUT32) out32[(size_t)m * Nout + n] = v;
          if (OUT16) out16[(size_t)m * Nout + n] = (_Float16)v;
        }
      }
    }
  }
}

// ---------------------------------------------------------------------------
// Flash attention: one block (128 thr = 4 waves) owns one (b,h) and 64 query
// rows (4 tiles of 16, one per wave).  Per 32-key block, K and V are staged
// in LDS once (coop b128 loads) and shared by all 4 waves; V is stored
// transposed so every B-fragment gather is two aligned 16B ds loads.
// qkv f16 layout: row m of [M,3C]: q at +0, k at +C, v at +2C, inner [H,DH].
// Output attn16 as [b,n,h,d] -> [M,C] f16 (ready for proj GEMM).
// ---------------------------------------------------------------------------
__global__ void attn_kernel(const _Float16* __restrict__ qkv,
                            _Float16* __restrict__ o16) {
  constexpr int NB64 = (TN + 63) / 64;      // 17 query blocks per (b,h)
  constexpr int KPAD = 8;

  __shared__ _Float16 Ks[32][TDH + KPAD];   // [key][dim]
  __shared__ _Float16 Vt[TDH][32 + KPAD];   // [dim][key]  (transposed)
  __shared__ _Float16 Plds[4][16 * 32];     // per-wave 16x32 P patch

  int tid  = threadIdx.x;
  int wave = tid >> 5;
  int lane = tid & 31;
  int hw   = lane >> 4;
  int l16  = lane & 15;

  int nb = blockIdx.x % NB64;
  int h  = (blockIdx.x / NB64) % TH;
  int b  = blockIdx.x / (NB64 * TH);
  int n0 = nb * 64 + wave * 16;             // this wave's query tile

  // ---- Q fragments (two 16x32 A-frags covering DH=64), rows clamped ----
  int qrow = n0 + l16; if (qrow >= TN) qrow = TN - 1;
  size_t qbase = ((size_t)(b * TN + qrow)) * T3C + h * TDH;
  v16h qf[2];
#pragma unroll
  for (int t2 = 0; t2 < 2; ++t2) {
    const _Float16* qp = qkv + qbase + t2 * 32 + hw * 16;
    h8 lo = *(const h8*)(qp);
    h8 hi = *(const h8*)(qp + 8);
#pragma unroll
    for (int i = 0; i < 8; ++i) { qf[t2][i] = lo[i]; qf[t2][8 + i] = hi[i]; }
  }

  float m_i[8], l_i[8];
#pragma unroll
  for (int j = 0; j < 8; ++j) { m_i[j] = -1e30f; l_i[j] = 0.f; }
  v8f Oacc[4] = {};

  for (int c0 = 0; c0 < TN; c0 += 32) {
    // ---- stage K (row-major) and V (transposed) tiles in LDS ----
#pragma unroll
    for (int c = 0; c < 2; ++c) {
      int cid = tid * 2 + c;                // 256 h8-chunks: 32 keys x 8
      int keyc = cid >> 3;
      int col  = (cid & 7) * 8;
      int kg   = c0 + keyc; if (kg >= TN) kg = TN - 1;
      size_t base = ((size_t)(b * TN + kg)) * T3C + h * TDH + col;
      h8 kv = *(const h8*)(qkv + base + TC);        // K
      h8 vv = *(const h8*)(qkv + base + 2 * TC);    // V
      *(h8*)&Ks[keyc][col] = kv;
#pragma unroll
      for (int i = 0; i < 8; ++i) Vt[col + i][keyc] = vv[i];
    }
    __syncthreads();

    // ---- scores for two 16-key tiles ----
    v8f S[2] = {};
#pragma unroll
    for (int t = 0; t < 2; ++t) {
      int kl = t * 16 + l16;                // local key (B-frag column)
#pragma unroll
      for (int t2 = 0; t2 < 2; ++t2) {
        const _Float16* kp = &Ks[kl][t2 * 32 + hw * 16];
        h8 lo = *(const h8*)(kp);
        h8 hi = *(const h8*)(kp + 8);
        v16h kf;
#pragma unroll
        for (int i = 0; i < 8; ++i) { kf[i] = lo[i]; kf[8 + i] = hi[i]; }
        S[t] = wmma_f16(qf[t2], kf, S[t]);
      }
    }
    // ---- scale + mask ----
#pragma unroll
    for (int t = 0; t < 2; ++t) {
      bool valid = (c0 + t * 16 + l16) < TN;
#pragma unroll
      for (int j = 0; j < 8; ++j)
        S[t][j] = valid ? S[t][j] * ATT_SCALE : -1e30f;
    }
    // ---- online softmax (row stats across 16-lane half-waves) ----
#pragma unroll
    for (int j = 0; j < 8; ++j) {
      float rm = fmaxf(S[0][j], S[1][j]);
#pragma unroll
      for (int o = 1; o < 16; o <<= 1) rm = fmaxf(rm, __shfl_xor(rm, o));
      float m_new = fmaxf(m_i[j], rm);
      float alpha = __expf(m_i[j] - m_new);
      float p0 = __expf(S[0][j] - m_new);
      float p1 = __expf(S[1][j] - m_new);
      float rs = p0 + p1;
#pragma unroll
      for (int o = 1; o < 16; o <<= 1) rs += __shfl_xor(rs, o);
      l_i[j] = l_i[j] * alpha + rs;
      m_i[j] = m_new;
#pragma unroll
      for (int d = 0; d < 4; ++d) Oacc[d][j] *= alpha;
      int prow = j + 8 * hw;
      Plds[wave][prow * 32 + l16]      = (_Float16)p0;
      Plds[wave][prow * 32 + 16 + l16] = (_Float16)p1;
    }
    asm volatile("s_wait_dscnt 0" ::: "memory");   // P stores visible (same wave)

    // ---- P (16x32 A-frag) x V (32x64 as 4 B-frags from transposed LDS) ----
    v16h pf;
    {
      const _Float16* pp = &Plds[wave][l16 * 32 + hw * 16];
      h8 lo = *(const h8*)(pp);
      h8 hi = *(const h8*)(pp + 8);
#pragma unroll
      for (int i = 0; i < 8; ++i) { pf[i] = lo[i]; pf[8 + i] = hi[i]; }
    }
#pragma unroll
    for (int d = 0; d < 4; ++d) {
      const _Float16* vp = &Vt[d * 16 + l16][hw * 16];
      h8 lo = *(const h8*)(vp);
      h8 hi = *(const h8*)(vp + 8);
      v16h vf;
#pragma unroll
      for (int i = 0; i < 8; ++i) { vf[i] = lo[i]; vf[8 + i] = hi[i]; }
      Oacc[d] = wmma_f16(pf, vf, Oacc[d]);
    }
    __syncthreads();                       // tiles consumed; safe to restage
  }

  // ---- normalize + store O as [b,n,h,d] f16 ----
#pragma unroll
  for (int j = 0; j < 8; ++j) {
    float inv = 1.0f / l_i[j];
    int qn = n0 + j + 8 * hw;
    if (qn < TN) {
#pragma unroll
      for (int d = 0; d < 4; ++d)
        o16[((size_t)(b * TN + qn)) * TC + h * TDH + d * 16 + l16] =
            (_Float16)(Oacc[d][j] * inv);
    }
  }
}

// ---------------------------------------------------------------------------
// Host-side orchestration (graph-capture safe).
// ---------------------------------------------------------------------------
static void* carve(char*& p, size_t bytes) {
  void* r = (void*)p;
  p += (bytes + 255) & ~(size_t)255;
  return r;
}

extern "C" void kernel_launch(void* const* d_in, const int* in_sizes, int n_in,
                              void* d_out, int out_size, void* d_ws, size_t ws_size,
                              hipStream_t stream) {
  const float* x_in   = (const float*)d_in[0];
  const float* ln1_w  = (const float*)d_in[1];
  const float* ln1_b  = (const float*)d_in[2];
  const float* qkv_w  = (const float*)d_in[3];
  const float* qkv_b  = (const float*)d_in[4];
  const float* proj_w = (const float*)d_in[5];
  const float* proj_b = (const float*)d_in[6];
  const float* ln2_w  = (const float*)d_in[7];
  const float* ln2_b  = (const float*)d_in[8];
  const float* fc1_w  = (const float*)d_in[9];
  const float* fc1_b  = (const float*)d_in[10];
  const float* fc2_w  = (const float*)d_in[11];
  const float* fc2_b  = (const float*)d_in[12];
  const float* norm_w = (const float*)d_in[13];
  const float* norm_b = (const float*)d_in[14];

  char* wp = (char*)d_ws;
  float*    xbuf   = (float*)   carve(wp, (size_t)TM * TC * sizeof(float));
  _Float16* h16    = (_Float16*)carve(wp, (size_t)TM * TC * sizeof(_Float16));
  _Float16* qkv16  = (_Float16*)carve(wp, (size_t)TM * T3C * sizeof(_Float16));
  _Float16* attn16 = (_Float16*)carve(wp, (size_t)TM * TC * sizeof(_Float16));
  _Float16* hid16  = (_Float16*)carve(wp, (size_t)TM * THID * sizeof(_Float16));
  _Float16* w16    = (_Float16*)carve(wp, (size_t)THID * TC * sizeof(_Float16));

  (void)hipMemcpyAsync(xbuf, x_in, (size_t)TM * TC * sizeof(float),
                       hipMemcpyDeviceToDevice, stream);

  dim3 tb256(256), tb128(128);
  dim3 gLN(TM);
  auto cvt = [&](const float* s, _Float16* d, int n) {
    int n4 = n / 4;
    cvt_f32_f16_kernel<<<(n4 + 255) / 256, tb256, 0, stream>>>(s, d, n4);
  };
  constexpr int GY = (TM + 127) / 128;             // 33
  constexpr int NB64 = (TN + 63) / 64;             // 17
  dim3 gAttn(TB * TH * NB64);                      // 1088 blocks

  for (int l = 0; l < TL; ++l) {
    // ---- attention ----
    ln_kernel<_Float16><<<gLN, tb256, 0, stream>>>(xbuf, ln1_w + l * TC,
                                                   ln1_b + l * TC, h16, TM);
    cvt(qkv_w + (size_t)l * T3C * TC, w16, T3C * TC);
    gemm_kernel<false, false, true, false><<<dim3(T3C / 128, GY), tb256, 0, stream>>>(
        h16, w16, qkv_b + l * T3C, nullptr, nullptr, qkv16, TM, T3C, TC);
    attn_kernel<<<gAttn, tb128, 0, stream>>>(qkv16, attn16);
    cvt(proj_w + (size_t)l * TC * TC, w16, TC * TC);
    gemm_kernel<false, true, false, true><<<dim3(TC / 128, GY), tb256, 0, stream>>>(
        attn16, w16, proj_b + l * TC, xbuf, xbuf, nullptr, TM, TC, TC);
    // ---- MLP ----
    ln_kernel<_Float16><<<gLN, tb256, 0, stream>>>(xbuf, ln2_w + l * TC,
                                                   ln2_b + l * TC, h16, TM);
    cvt(fc1_w + (size_t)l * THID * TC, w16, THID * TC);
    gemm_kernel<true, false, true, false><<<dim3(THID / 128, GY), tb256, 0, stream>>>(
        h16, w16, fc1_b + l * THID, nullptr, nullptr, hid16, TM, THID, TC);
    cvt(fc2_w + (size_t)l * TC * THID, w16, TC * THID);
    gemm_kernel<false, true, false, true><<<dim3(TC / 128, GY), tb256, 0, stream>>>(
        hid16, w16, fc2_b + l * TC, xbuf, xbuf, nullptr, TM, TC, THID);
  }

  ln_kernel<float><<<gLN, tb256, 0, stream>>>(xbuf, norm_w, norm_b,
                                              (float*)d_out, TM);
}
